// EnhancedGNNEncoder_12249246728343
// MI455X (gfx1250) — matrix operand order repaired and checked by reading, not compiled, not run on hardware
//
#include <hip/hip_runtime.h>
#include <hip/hip_bf16.h>
#include <string.h>

// ---------------------------------------------------------------------------
// EnhancedGNNEncoder for MI455X (gfx1250, wave32, WMMA).
// All dense math runs through V_WMMA_F32_16X16X32_BF16; graph scatter ops use
// f32 global atomics; per-graph MHA is a flash-attention kernel with online
// softmax staged through LDS (s_wait_dscnt) between the two WMMA stages.
// GEMM: one 16(M)x64(N) strip per wave (4 WMMA accumulators); all 4 B tiles
// of a k-step are loaded into distinct registers BEFORE the WMMA chain so the
// backend can clause the loads and use partial s_wait_loadcnt.
// ---------------------------------------------------------------------------

typedef __bf16 bf16;
typedef __attribute__((ext_vector_type(16))) __bf16 v16bf;
typedef __attribute__((ext_vector_type(8)))  __bf16 v8bf;
typedef __attribute__((ext_vector_type(8)))  float  v8f;

#define NN      65536      // nodes
#define NE      524288     // edges (without self loops)
#define GS      512        // graph size
#define NG      128        // graphs
#define H4      4          // heads (layers 1,2)
#define C64     64         // channels/head (layers 1,2)
#define HC      256        // 4*64
#define OD      128        // out dim
#define HD      32         // MHA head dim

// ------------------------------ helpers ------------------------------------

__device__ __forceinline__ float lrelu02(float x) { return x > 0.f ? x : 0.2f * x; }
// order-preserving float<->uint mapping for atomicMax on floats
__device__ __forceinline__ unsigned encf(float f) {
  unsigned b = __float_as_uint(f);
  return (b & 0x80000000u) ? ~b : (b | 0x80000000u);
}
__device__ __forceinline__ float decf(unsigned u) {
  return (u & 0x80000000u) ? __uint_as_float(u ^ 0x80000000u) : __uint_as_float(~u);
}
__device__ __forceinline__ void edge_sd(const int* __restrict__ ei, int E, int idx,
                                        int& src, int& dst) {
  if (idx < E) { src = ei[idx]; dst = ei[E + idx]; }
  else         { src = idx - E; dst = idx - E; }   // self loops appended
}
__device__ __forceinline__ v16bf cat16(v8bf lo, v8bf hi) {
  v16bf r;
#pragma unroll
  for (int i = 0; i < 8; ++i) { r[i] = lo[i]; r[i + 8] = hi[i]; }
  return r;
}

// ------------------------------ fills / converts ---------------------------

__global__ void fill_f32(float* p, float v, int n) {
  int t = blockIdx.x * blockDim.x + threadIdx.x;
  if (t < n) p[t] = v;
}
__global__ void fill_u32(unsigned* p, unsigned v, int n) {
  int t = blockIdx.x * blockDim.x + threadIdx.x;
  if (t < n) p[t] = v;
}
__global__ void cvt_bf16(const float* __restrict__ in, bf16* __restrict__ out, int n) {
  int t = blockIdx.x * blockDim.x + threadIdx.x;
  if (t < n) out[t] = (bf16)in[t];
}
// out[c*R + r] = in[r*C + c]   (weights: [K,N] f32 -> BT [N,K] bf16)
__global__ void cvt_transpose_bf16(const float* __restrict__ in, bf16* __restrict__ out,
                                   int R, int C) {
  int t = blockIdx.x * blockDim.x + threadIdx.x;
  if (t >= R * C) return;
  int r = t / C, c = t % C;
  out[(size_t)c * R + r] = (bf16)in[t];
}
// V [NN,128] f32 -> VT [NG*H4][HD][GS] bf16  (per graph/head transposed)
__global__ void v_transpose_bf16(const float* __restrict__ V, bf16* __restrict__ VT) {
  int t = blockIdx.x * blockDim.x + threadIdx.x;
  if (t >= NN * OD) return;
  int i = t >> 7, c = t & 127;
  int g = i >> 9, j = i & 511;
  int h = c >> 5, d = c & 31;
  VT[(((size_t)(g * H4 + h) * HD) + d) * GS + j] = (bf16)V[t];
}

// ------------------------------ WMMA GEMM ----------------------------------
// C[M,N] = A[M,K] (bf16, row-major, lda) * B (given as BT[N,K] bf16, ldbt)
//          + bias[N] (optional).  One 16x64 strip per wave: 4 accumulators,
// A loaded once per 32-wide k-step and reused across 4 B tiles.  All B tiles
// are loaded into distinct live registers before the WMMA chain (clauseable).
// A lane layout: lane L (r=L&15, hi=L>>4) holds
//   K = hi*16+{0..15} of row r (same k order as B).
// B lane layout (same k order as A): lane L holds K = hi*16+{0..15} of col r.
__global__ void __launch_bounds__(128)
gemm_bf16_wmma(const bf16* __restrict__ A, const bf16* __restrict__ BT,
               float* __restrict__ C, const float* __restrict__ bias,
               int M, int N, int K, int lda, int ldbt, int ldc) {
  const int tilesM = M >> 4, stripsN = N >> 6;         // N % 64 == 0
  int wid = blockIdx.x * 4 + (threadIdx.x >> 5);
  if (wid >= tilesM * stripsN) return;                 // wave-uniform guard
  int tm = wid % tilesM, tn = wid / tilesM;
  int lane = threadIdx.x & 31, r = lane & 15, hi = lane >> 4;

  const bf16* arow = A + (size_t)(tm * 16 + r) * lda;
  const bf16* b0 = BT + (size_t)(tn * 64 +  0 + r) * ldbt;
  const bf16* b1 = BT + (size_t)(tn * 64 + 16 + r) * ldbt;
  const bf16* b2 = BT + (size_t)(tn * 64 + 32 + r) * ldbt;
  const bf16* b3 = BT + (size_t)(tn * 64 + 48 + r) * ldbt;

  v8f acc0 = {}, acc1 = {}, acc2 = {}, acc3 = {};
  for (int k0 = 0; k0 < K; k0 += 32) {
    // issue all loads first: 2 A halves + 4x2 B halves (distinct registers)
    v8bf alo = *(const v8bf*)(arow + k0 + hi * 16);
    v8bf ahi = *(const v8bf*)(arow + k0 + hi * 16 + 8);
    v8bf b0l = *(const v8bf*)(b0 + k0 + hi * 16);
    v8bf b0h = *(const v8bf*)(b0 + k0 + hi * 16 + 8);
    v8bf b1l = *(const v8bf*)(b1 + k0 + hi * 16);
    v8bf b1h = *(const v8bf*)(b1 + k0 + hi * 16 + 8);
    v8bf b2l = *(const v8bf*)(b2 + k0 + hi * 16);
    v8bf b2h = *(const v8bf*)(b2 + k0 + hi * 16 + 8);
    v8bf b3l = *(const v8bf*)(b3 + k0 + hi * 16);
    v8bf b3h = *(const v8bf*)(b3 + k0 + hi * 16 + 8);
    v16bf a = cat16(alo, ahi);
    acc0 = __builtin_amdgcn_wmma_f32_16x16x32_bf16(false, a, false, cat16(b0l, b0h),
                                                   (short)0, acc0, false, false);
    acc1 = __builtin_amdgcn_wmma_f32_16x16x32_bf16(false, a, false, cat16(b1l, b1h),
                                                   (short)0, acc1, false, false);
    acc2 = __builtin_amdgcn_wmma_f32_16x16x32_bf16(false, a, false, cat16(b2l, b2h),
                                                   (short)0, acc2, false, false);
    acc3 = __builtin_amdgcn_wmma_f32_16x16x32_bf16(false, a, false, cat16(b3l, b3h),
                                                   (short)0, acc3, false, false);
  }
  v8f accs[4] = {acc0, acc1, acc2, acc3};
#pragma unroll
  for (int s = 0; s < 4; ++s) {
    int col = tn * 64 + s * 16 + r;
    float bv = bias ? bias[col] : 0.f;
#pragma unroll
    for (int v = 0; v < 8; ++v) {                      // C layout: row = v+8*hi
      int row = tm * 16 + v + 8 * hi;
      C[(size_t)row * ldc + col] = accs[s][v] + bv;
    }
  }
}

// ------------------------------ GAT pieces ---------------------------------

// e_s[i,h] = sum_c h[i,h,c]*a_s[h,c] ; e_d likewise
__global__ void attn_logits(const float* __restrict__ lin, const float* __restrict__ as_,
                            const float* __restrict__ ad_, float* __restrict__ es,
                            float* __restrict__ ed, int N, int H, int C) {
  int t = blockIdx.x * blockDim.x + threadIdx.x;
  if (t >= N * H) return;
  int i = t / H, h = t % H;
  const float* hp = lin + (size_t)i * H * C + h * C;
  float ss = 0.f, sd = 0.f;
  for (int c = 0; c < C; ++c) { float v = hp[c]; ss += v * as_[h * C + c]; sd += v * ad_[h * C + c]; }
  es[t] = ss; ed[t] = sd;
}

__global__ void edge_max(const int* __restrict__ ei, const float* __restrict__ es,
                         const float* __restrict__ ed, unsigned* __restrict__ m,
                         int E, int N, int H) {
  int t = blockIdx.x * blockDim.x + threadIdx.x;
  if (t >= (E + N) * H) return;
  int e = t / H, h = t % H, src, dst;
  edge_sd(ei, E, e, src, dst);
  float v = lrelu02(es[src * H + h] + ed[dst * H + h]);
  atomicMax(&m[dst * H + h], encf(v));
}

__global__ void edge_sum(const int* __restrict__ ei, const float* __restrict__ es,
                         const float* __restrict__ ed, const unsigned* __restrict__ m,
                         float* __restrict__ z, int E, int N, int H) {
  int t = blockIdx.x * blockDim.x + threadIdx.x;
  if (t >= (E + N) * H) return;
  int e = t / H, h = t % H, src, dst;
  edge_sd(ei, E, e, src, dst);
  float v = lrelu02(es[src * H + h] + ed[dst * H + h]);
  atomicAdd(&z[dst * H + h], __expf(v - decf(m[dst * H + h])));
}

__global__ void edge_aggr(const int* __restrict__ ei, const float* __restrict__ es,
                          const float* __restrict__ ed, const unsigned* __restrict__ m,
                          const float* __restrict__ z, const float* __restrict__ lin,
                          float* __restrict__ agg, int E, int N, int H, int C) {
  int t = blockIdx.x * blockDim.x + threadIdx.x;
  if (t >= (E + N) * H) return;
  int e = t / H, h = t % H, src, dst;
  edge_sd(ei, E, e, src, dst);
  float v = lrelu02(es[src * H + h] + ed[dst * H + h]);
  float p = __expf(v - decf(m[dst * H + h]));
  float alpha = p / (z[dst * H + h] + 1e-16f);
  const float* hs = lin + (size_t)src * H * C + h * C;
  float* ao = agg + (size_t)dst * H * C + h * C;
  for (int c = 0; c < C; ++c) atomicAdd(&ao[c], alpha * hs[c]);
}

// out_bf16 = LayerNorm(elu(agg + bias)) * gamma + beta  (256 ch, 1 wave/node)
__global__ void __launch_bounds__(128)
elu_ln_bf16(const float* __restrict__ agg, const float* __restrict__ bias,
            const float* __restrict__ gam, const float* __restrict__ bet,
            bf16* __restrict__ out) {
  int node = blockIdx.x * 4 + (threadIdx.x >> 5);
  int lane = threadIdx.x & 31;
  const float* row = agg + (size_t)node * HC;
  float v[8], s = 0.f;
#pragma unroll
  for (int j = 0; j < 8; ++j) {
    int c = lane + 32 * j;
    float x = row[c] + bias[c];
    x = x > 0.f ? x : (__expf(x) - 1.f);               // elu
    v[j] = x; s += x;
  }
#pragma unroll
  for (int msk = 1; msk < 32; msk <<= 1) s += __shfl_xor(s, msk, 32);
  float mu = s * (1.f / 256.f), var = 0.f;
#pragma unroll
  for (int j = 0; j < 8; ++j) { float d = v[j] - mu; var += d * d; }
#pragma unroll
  for (int msk = 1; msk < 32; msk <<= 1) var += __shfl_xor(var, msk, 32);
  float inv = rsqrtf(var * (1.f / 256.f) + 1e-5f);
#pragma unroll
  for (int j = 0; j < 8; ++j) {
    int c = lane + 32 * j;
    out[(size_t)node * HC + c] = (bf16)((v[j] - mu) * inv * gam[c] + bet[c]);
  }
}

// h3_bf16 = agg3 + b3 + resid(=x@Wr+br already)   (layer-3 epilogue)
__global__ void add3_bf16(const float* __restrict__ agg, const float* __restrict__ b3,
                          const float* __restrict__ resid, bf16* __restrict__ out, int n) {
  int t = blockIdx.x * blockDim.x + threadIdx.x;
  if (t >= n) return;
  out[t] = (bf16)(agg[t] + b3[t & 127] + resid[t]);
}

// ------------------------------ flash attention ----------------------------
// One wave per (graph g, head h, 16-query-row tile qt). hd=32 => QK^T is a
// single WMMA K-step; keys processed in pairs of 16 so P forms a full 16x32
// WMMA-A operand (staged via LDS); P@V = two WMMAs (dims 0-15 / 16-31).
__global__ void __launch_bounds__(128)
flash_attn(const bf16* __restrict__ Qb, const bf16* __restrict__ Kb,
           const bf16* __restrict__ VT, float* __restrict__ O, float scale) {
  __shared__ bf16 pbuf[4][16][32];
  int wslot = threadIdx.x >> 5;
  int wid = blockIdx.x * 4 + wslot;
  int qt = wid & 31;
  int gh = wid >> 5;
  int g = gh >> 2, h = gh & 3;
  int lane = threadIdx.x & 31, r = lane & 15, hi = lane >> 4;

  // Q tile in WMMA-A layout (row r, K = head-dim)
  const bf16* qrow = Qb + (size_t)(g * GS + qt * 16 + r) * OD + h * HD;
  v16bf aq = cat16(*(const v8bf*)(qrow + hi * 16), *(const v8bf*)(qrow + hi * 16 + 8));
  const bf16* kbase = Kb + (size_t)(g * GS) * OD + h * HD;
  const bf16* vtb   = VT + (size_t)gh * HD * GS;        // [32][512]

  float rm[8], rs[8];
  v8f acc0 = {}, acc1 = {};
#pragma unroll
  for (int v = 0; v < 8; ++v) { rm[v] = -3.0e38f; rs[v] = 0.f; }

  for (int pair = 0; pair < GS / 32; ++pair) {
#pragma unroll
    for (int tkt = 0; tkt < 2; ++tkt) {
      int kt = pair * 2 + tkt;
      const bf16* krow = kbase + (size_t)(kt * 16 + r) * OD;   // BT row = key j
      v16bf bk = cat16(*(const v8bf*)(krow + hi * 16),
                       *(const v8bf*)(krow + hi * 16 + 8));
      v8f zero = {};
      v8f e = __builtin_amdgcn_wmma_f32_16x16x32_bf16(false, aq, false, bk,
                                                      (short)0, zero, false, false);
      // online softmax; C layout: elem v -> (q-row v+8*hi, key col kt*16+r)
#pragma unroll
      for (int v = 0; v < 8; ++v) {
        float ev = e[v] * scale;
        float tmax = ev;
#pragma unroll
        for (int msk = 1; msk < 16; msk <<= 1) tmax = fmaxf(tmax, __shfl_xor(tmax, msk, 32));
        float nm = fmaxf(rm[v], tmax);
        float p = __expf(ev - nm);
        float tsum = p;
#pragma unroll
        for (int msk = 1; msk < 16; msk <<= 1) tsum += __shfl_xor(tsum, msk, 32);
        float corr = __expf(rm[v] - nm);
        rs[v] = rs[v] * corr + tsum;
        rm[v] = nm;
        acc0[v] *= corr; acc1[v] *= corr;
        pbuf[wslot][v + 8 * hi][tkt * 16 + r] = (bf16)p;
      }
    }
    asm volatile("s_wait_dscnt 0" ::: "memory");        // LDS writes visible
    // reload P as WMMA-A (row r, K = 32 keys of this pair)
    v16bf ap;
#pragma unroll
    for (int i = 0; i < 8; ++i) {
      ap[i]     = pbuf[wslot][r][hi * 16 + i];
      ap[i + 8] = pbuf[wslot][r][hi * 16 + 8 + i];
    }
    // V^T tiles: BT row = out dim d, K = key index (contiguous in VT)
    const bf16* v0 = vtb + (size_t)r * GS + pair * 32;
    const bf16* v1 = vtb + (size_t)(16 + r) * GS + pair * 32;
    v16bf bv0 = cat16(*(const v8bf*)(v0 + hi * 16), *(const v8bf*)(v0 + hi * 16 + 8));
    v16bf bv1 = cat16(*(const v8bf*)(v1 + hi * 16), *(const v8bf*)(v1 + hi * 16 + 8));
    acc0 = __builtin_amdgcn_wmma_f32_16x16x32_bf16(false, ap, false, bv0,
                                                   (short)0, acc0, false, false);
    acc1 = __builtin_amdgcn_wmma_f32_16x16x32_bf16(false, ap, false, bv1,
                                                   (short)0, acc1, false, false);
  }
#pragma unroll
  for (int v = 0; v < 8; ++v) {
    int row = qt * 16 + v + 8 * hi;
    float inv = 1.f / rs[v];
    float* orow = O + (size_t)(g * GS + row) * OD + h * HD;
    orow[r]      = acc0[v] * inv;
    orow[16 + r] = acc1[v] * inv;
  }
}

// ------------------------------ launcher -----------------------------------

extern "C" void kernel_launch(void* const* d_in, const int* in_sizes, int n_in,
                              void* d_out, int out_size, void* d_ws, size_t ws_size,
                              hipStream_t stream) {
  const float* x   = (const float*)d_in[0];
  const int*   ei  = (const int*)d_in[1];
  const float* W1  = (const float*)d_in[3];
  const float* a1s = (const float*)d_in[4];
  const float* a1d = (const float*)d_in[5];
  const float* b1  = (const float*)d_in[6];
  const float* W2  = (const float*)d_in[7];
  const float* a2s = (const float*)d_in[8];
  const float* a2d = (const float*)d_in[9];
  const float* b2  = (const float*)d_in[10];
  const float* W3  = (const float*)d_in[11];
  const float* a3s = (const float*)d_in[12];
  const float* a3d = (const float*)d_in[13];
  const float* b3  = (const float*)d_in[14];
  const float* g1  = (const float*)d_in[15];
  const float* be1 = (const float*)d_in[16];
  const float* g2  = (const float*)d_in[17];
  const float* be2 = (const float*)d_in[18];
  const float* Wr  = (const float*)d_in[19];
  const float* br  = (const float*)d_in[20];
  const float* Wq  = (const float*)d_in[21];
  const float* Wk  = (const float*)d_in[22];
  const float* Wv  = (const float*)d_in[23];
  const float* Wo  = (const float*)d_in[24];
  const float* bo  = (const float*)d_in[25];
  float* out = (float*)d_out;
  const int E = in_sizes[1] / 2;

  // workspace layout
  char* p = (char*)d_ws;
  auto alloc = [&](size_t bytes) { void* q = p; p += (bytes + 255) & ~(size_t)255; return q; };
  bf16* xbf  = (bf16*)alloc((size_t)NN * 64 * 2);
  bf16* hbf  = (bf16*)alloc((size_t)NN * HC * 2);      // bf16 activations (reused)
  bf16* W1t  = (bf16*)alloc(256 * 64 * 2);
  bf16* W2t  = (bf16*)alloc(256 * 256 * 2);
  bf16* W3t  = (bf16*)alloc(128 * 256 * 2);
  bf16* Wrt  = (bf16*)alloc(128 * 64 * 2);
  bf16* Wqt  = (bf16*)alloc(128 * 128 * 2);
  bf16* Wkt  = (bf16*)alloc(128 * 128 * 2);
  bf16* Wvt  = (bf16*)alloc(128 * 128 * 2);
  bf16* Wot  = (bf16*)alloc(128 * 128 * 2);
  float* lin = (float*)alloc((size_t)NN * HC * 4);     // GEMM out; later Qf|Kf
  float* agg = (float*)alloc((size_t)NN * HC * 4);     // aggregation; later Vf|Of
  float* es  = (float*)alloc((size_t)NN * H4 * 4);
  float* ed  = (float*)alloc((size_t)NN * H4 * 4);
  unsigned* mx = (unsigned*)alloc((size_t)NN * H4 * 4);
  float* zz  = (float*)alloc((size_t)NN * H4 * 4);
  float* resid = (float*)alloc((size_t)NN * OD * 4);
  bf16* Qbf  = (bf16*)alloc((size_t)NN * OD * 2);
  bf16* Kbf  = (bf16*)alloc((size_t)NN * OD * 2);
  bf16* VTb  = (bf16*)alloc((size_t)NG * H4 * HD * GS * 2);
  bf16* Obf  = (bf16*)alloc((size_t)NN * OD * 2);
  float* Qf = lin;            float* Kf = lin + (size_t)NN * OD;
  float* Vf = agg;            float* Of = agg + (size_t)NN * OD;

  float negInf = -3.0e38f; unsigned encInit; memcpy(&encInit, &negInf, 4); encInit = ~encInit;
  auto grid1 = [](int n) { return dim3((unsigned)((n + 255) / 256)); };
  auto gemmGrid = [](int M, int N) { return dim3((unsigned)(((M >> 4) * (N >> 6) + 3) / 4)); };

  // weight prep
  cvt_transpose_bf16<<<grid1(64 * 256), 256, 0, stream>>>(W1, W1t, 64, 256);
  cvt_transpose_bf16<<<grid1(256 * 256), 256, 0, stream>>>(W2, W2t, 256, 256);
  cvt_transpose_bf16<<<grid1(256 * 128), 256, 0, stream>>>(W3, W3t, 256, 128);
  cvt_transpose_bf16<<<grid1(64 * 128), 256, 0, stream>>>(Wr, Wrt, 64, 128);
  cvt_transpose_bf16<<<grid1(128 * 128), 256, 0, stream>>>(Wq, Wqt, 128, 128);
  cvt_transpose_bf16<<<grid1(128 * 128), 256, 0, stream>>>(Wk, Wkt, 128, 128);
  cvt_transpose_bf16<<<grid1(128 * 128), 256, 0, stream>>>(Wv, Wvt, 128, 128);
  cvt_transpose_bf16<<<grid1(128 * 128), 256, 0, stream>>>(Wo, Wot, 128, 128);
  cvt_bf16<<<grid1(NN * 64), 256, 0, stream>>>(x, xbf, NN * 64);

  const int ET4 = (E + NN) * H4, ET1 = (E + NN);

  // ---- GAT layer 1 (H=4, C=64) ----
  gemm_bf16_wmma<<<gemmGrid(NN, HC), 128, 0, stream>>>(xbf, W1t, lin, nullptr, NN, HC, 64, 64, 64, HC);
  attn_logits<<<grid1(NN * H4), 256, 0, stream>>>(lin, a1s, a1d, es, ed, NN, H4, C64);
  fill_u32<<<grid1(NN * H4), 256, 0, stream>>>(mx, encInit, NN * H4);
  fill_f32<<<grid1(NN * H4), 256, 0, stream>>>(zz, 0.f, NN * H4);
  fill_f32<<<grid1(NN * HC), 256, 0, stream>>>(agg, 0.f, NN * HC);
  edge_max<<<grid1(ET4), 256, 0, stream>>>(ei, es, ed, mx, E, NN, H4);
  edge_sum<<<grid1(ET4), 256, 0, stream>>>(ei, es, ed, mx, zz, E, NN, H4);
  edge_aggr<<<grid1(ET4), 256, 0, stream>>>(ei, es, ed, mx, zz, lin, agg, E, NN, H4, C64);
  elu_ln_bf16<<<NN / 4, 128, 0, stream>>>(agg, b1, g1, be1, hbf);

  // ---- GAT layer 2 (H=4, C=64) ----
  gemm_bf16_wmma<<<gemmGrid(NN, HC), 128, 0, stream>>>(hbf, W2t, lin, nullptr, NN, HC, HC, HC, HC, HC);
  attn_logits<<<grid1(NN * H4), 256, 0, stream>>>(lin, a2s, a2d, es, ed, NN, H4, C64);
  fill_u32<<<grid1(NN * H4), 256, 0, stream>>>(mx, encInit, NN * H4);
  fill_f32<<<grid1(NN * H4), 256, 0, stream>>>(zz, 0.f, NN * H4);
  fill_f32<<<grid1(NN * HC), 256, 0, stream>>>(agg, 0.f, NN * HC);
  edge_max<<<grid1(ET4), 256, 0, stream>>>(ei, es, ed, mx, E, NN, H4);
  edge_sum<<<grid1(ET4), 256, 0, stream>>>(ei, es, ed, mx, zz, E, NN, H4);
  edge_aggr<<<grid1(ET4), 256, 0, stream>>>(ei, es, ed, mx, zz, lin, agg, E, NN, H4, C64);
  elu_ln_bf16<<<NN / 4, 128, 0, stream>>>(agg, b2, g2, be2, hbf);

  // ---- GAT layer 3 (H=1, C=128) ----
  gemm_bf16_wmma<<<gemmGrid(NN, OD), 128, 0, stream>>>(hbf, W3t, lin, nullptr, NN, OD, HC, HC, HC, OD);
  attn_logits<<<grid1(NN), 256, 0, stream>>>(lin, a3s, a3d, es, ed, NN, 1, OD);
  fill_u32<<<grid1(NN), 256, 0, stream>>>(mx, encInit, NN);
  fill_f32<<<grid1(NN), 256, 0, stream>>>(zz, 0.f, NN);
  fill_f32<<<grid1(NN * OD), 256, 0, stream>>>(agg, 0.f, NN * OD);
  edge_max<<<grid1(ET1), 256, 0, stream>>>(ei, es, ed, mx, E, NN, 1);
  edge_sum<<<grid1(ET1), 256, 0, stream>>>(ei, es, ed, mx, zz, E, NN, 1);
  edge_aggr<<<grid1(ET1), 256, 0, stream>>>(ei, es, ed, mx, zz, lin, agg, E, NN, 1, OD);
  // residual x@Wr + br  (bias fused in GEMM)
  gemm_bf16_wmma<<<gemmGrid(NN, OD), 128, 0, stream>>>(xbf, Wrt, resid, br, NN, OD, 64, 64, 64, OD);
  add3_bf16<<<grid1(NN * OD), 256, 0, stream>>>(agg, b3, resid, hbf, NN * OD);

  // ---- per-graph MHA ----
  gemm_bf16_wmma<<<gemmGrid(NN, OD), 128, 0, stream>>>(hbf, Wqt, Qf, nullptr, NN, OD, OD, OD, OD, OD);
  gemm_bf16_wmma<<<gemmGrid(NN, OD), 128, 0, stream>>>(hbf, Wkt, Kf, nullptr, NN, OD, OD, OD, OD, OD);
  gemm_bf16_wmma<<<gemmGrid(NN, OD), 128, 0, stream>>>(hbf, Wvt, Vf, nullptr, NN, OD, OD, OD, OD, OD);
  cvt_bf16<<<grid1(NN * OD), 256, 0, stream>>>(Qf, Qbf, NN * OD);
  cvt_bf16<<<grid1(NN * OD), 256, 0, stream>>>(Kf, Kbf, NN * OD);
  v_transpose_bf16<<<grid1(NN * OD), 256, 0, stream>>>(Vf, VTb);
  flash_attn<<<NG * H4 * (GS / 16) / 4, 128, 0, stream>>>(Qbf, Kbf, VTb, Of,
                                                          1.f / sqrtf((float)OD));
  cvt_bf16<<<grid1(NN * OD), 256, 0, stream>>>(Of, Obf, NN * OD);
  gemm_bf16_wmma<<<gemmGrid(NN, OD), 128, 0, stream>>>(Obf, Wot, out, bo, NN, OD, OD, OD, OD, OD);
}
